// Llama4TextExperts_53798760349861
// MI455X (gfx1250) — matrix operand order repaired and checked by reading, not compiled, not run on hardware
//
#include <hip/hip_runtime.h>
#include <hip/hip_bf16.h>

// ---------------------------------------------------------------------------
// MoE grouped FFN for MI455X (gfx1250, wave32):
//   hidden = up * silu(gate),  gate_up = X @ Wgu ;  out = hidden @ Wd
// bf16 WMMA (v_wmma_f32_16x16x32_bf16), fp32 accumulate, double-buffered LDS,
// exact v_cvt_pk_bf16_f32 staging, async global->LDS copies (ASYNCcnt) for
// the bf16 intermediate in GEMM2.
// ---------------------------------------------------------------------------

typedef __attribute__((ext_vector_type(8)))  __bf16 v8bf;
typedef __attribute__((ext_vector_type(16))) __bf16 v16bf;
typedef __attribute__((ext_vector_type(8)))  float  v8f;

constexpr int E  = 8;
constexpr int H  = 2048;   // hidden size (K of GEMM1, N of GEMM2)
constexpr int F  = 4096;   // expert dim  (N of GEMM1, K of GEMM2)
constexpr int TE = 1024;   // tokens per expert

constexpr int BM  = 128;    // block tile M
constexpr int BN  = 64;     // block tile N
constexpr int BK  = 32;     // K step (= bf16 WMMA depth)
constexpr int LDA = BK + 8; // padded LDS row (80 B) -> conflict-free banks

// Exact packed conversion: one v_cvt_pk_bf16_f32, no SLP repacking.
__device__ __forceinline__ unsigned pk2(float lo, float hi) {
    unsigned r;
    asm("v_cvt_pk_bf16_f32 %0, %1, %2" : "=v"(r) : "v"(lo), "v"(hi));
    return r;
}

// LDS byte offset of a __shared__ object (AS3 ptrtoint == offset in LDS).
__device__ __forceinline__ unsigned lds_off(const void* p) {
    return (unsigned)(unsigned long long)
        (__attribute__((address_space(3))) const char*)p;
}

// Per-lane async copy: 16 B global -> LDS, tracked by ASYNCcnt.
__device__ __forceinline__ void async_b128(unsigned dst_lds, const void* src) {
    asm volatile("global_load_async_to_lds_b128 %0, %1, off"
                 :: "v"(dst_lds), "v"(src) : "memory");
}
__device__ __forceinline__ void wait_async0() {
    asm volatile("s_wait_asynccnt 0" ::: "memory");
}

// Build a v16bf fragment from two contiguous 16B LDS chunks.
__device__ __forceinline__ v16bf frag_from(const __bf16* p0, const __bf16* p1) {
    v8bf lo = *(const v8bf*)p0;
    v8bf hi = *(const v8bf*)p1;
    return __builtin_shufflevector(lo, hi, 0,1,2,3,4,5,6,7,8,9,10,11,12,13,14,15);
}

// ---------------------------------------------------------------------------
// Kernel 1: gate_up GEMM + SwiGLU epilogue -> bf16 hidden in workspace
//   X   : [E][TE][H] fp32 ; Wgu : [E][H][2F] fp32 ; Hid : [E][TE][F] bf16
// ---------------------------------------------------------------------------
__global__ __launch_bounds__(256) void
moe_gateup_swiglu(const float* __restrict__ X,
                  const float* __restrict__ Wgu,
                  __bf16* __restrict__ Hid) {
    __shared__ __bf16 lA [2][BM][LDA];
    __shared__ __bf16 lBg[2][BN][LDA];
    __shared__ __bf16 lBu[2][BN][LDA];

    const int e  = blockIdx.z;
    const int m0 = blockIdx.y * BM;
    const int n0 = blockIdx.x * BN;          // column offset in F-space

    const float* Ae = X   + (size_t)e * TE * H;
    const float* We = Wgu + (size_t)e * H * (2 * F);

    const int tid  = threadIdx.x;
    const int lane = tid & 31;
    const int w    = tid >> 5;
    const int wr   = w >> 1;                 // wave row 0..3 (M)
    const int wc   = w & 1;                  // wave col 0..1 (N)

    const v8f zero = {0.f,0.f,0.f,0.f,0.f,0.f,0.f,0.f};
    v8f accG[2][2] = {{zero, zero}, {zero, zero}};
    v8f accU[2][2] = {{zero, zero}, {zero, zero}};

    // per-lane fragment addressing (ISA A/B layouts for 16x16x32 bf16)
    const int lr   = lane & 15;
    const int sel  = lane >> 4;
    const int aoff = sel * 8;                // A: chunks at {aoff, aoff+16}
    const int boff = sel * 8;                // B: chunks at {boff, boff+16}

    // global-load thread mapping
    const int arow = tid >> 1;               // A: 128 rows x 32 K, 16 f32/thread
    const int acb  = (tid & 1) * 16;
    const int bkp  = tid >> 4;               // B: K-pair index 0..15 -> k = 2*bkp
    const int bnb  = (tid & 15) * 4;         //    4 columns per thread

    const float* srcA = Ae + (size_t)(m0 + arow) * H + acb;
    const float* srcG = We + (size_t)(2 * bkp) * (2 * F) + n0 + bnb;
    const float* srcU = srcG + F;
    const size_t bstep = (size_t)BK * (2 * F);

    // ---- prologue: stage K-step 0 into buffer 0 ----
    {
        float4 a0 = *(const float4*)(srcA + 0);
        float4 a1 = *(const float4*)(srcA + 4);
        float4 a2 = *(const float4*)(srcA + 8);
        float4 a3 = *(const float4*)(srcA + 12);
        float4 g0 = *(const float4*)(srcG);
        float4 g1 = *(const float4*)(srcG + 2 * F);
        float4 u0 = *(const float4*)(srcU);
        float4 u1 = *(const float4*)(srcU + 2 * F);
        uint4 w0 = {pk2(a0.x,a0.y), pk2(a0.z,a0.w), pk2(a1.x,a1.y), pk2(a1.z,a1.w)};
        uint4 w1 = {pk2(a2.x,a2.y), pk2(a2.z,a2.w), pk2(a3.x,a3.y), pk2(a3.z,a3.w)};
        *(uint4*)&lA[0][arow][acb + 0] = w0;
        *(uint4*)&lA[0][arow][acb + 8] = w1;
        const int kk = 2 * bkp;
        *(unsigned*)&lBg[0][bnb + 0][kk] = pk2(g0.x, g1.x);
        *(unsigned*)&lBg[0][bnb + 1][kk] = pk2(g0.y, g1.y);
        *(unsigned*)&lBg[0][bnb + 2][kk] = pk2(g0.z, g1.z);
        *(unsigned*)&lBg[0][bnb + 3][kk] = pk2(g0.w, g1.w);
        *(unsigned*)&lBu[0][bnb + 0][kk] = pk2(u0.x, u1.x);
        *(unsigned*)&lBu[0][bnb + 1][kk] = pk2(u0.y, u1.y);
        *(unsigned*)&lBu[0][bnb + 2][kk] = pk2(u0.z, u1.z);
        *(unsigned*)&lBu[0][bnb + 3][kk] = pk2(u0.w, u1.w);
    }
    __syncthreads();

    constexpr int NSTEP = H / BK;            // 64
    for (int s = 0; s < NSTEP; ++s) {
        const int cur = s & 1;
        const bool has_next = (s + 1) < NSTEP;

        // 1) issue next-step global loads early (latency hidden under WMMA)
        float4 a0, a1, a2, a3, g0, g1, u0, u1;
        if (has_next) {
            const float* pa = srcA + (s + 1) * BK;
            const float* pg = srcG + (s + 1) * bstep;
            const float* pu = srcU + (s + 1) * bstep;
            a0 = *(const float4*)(pa + 0);
            a1 = *(const float4*)(pa + 4);
            a2 = *(const float4*)(pa + 8);
            a3 = *(const float4*)(pa + 12);
            g0 = *(const float4*)(pg);
            g1 = *(const float4*)(pg + 2 * F);
            u0 = *(const float4*)(pu);
            u1 = *(const float4*)(pu + 2 * F);
            if (s + 2 < NSTEP) {
                __builtin_prefetch(pa + BK, 0, 1);
                __builtin_prefetch(pg + bstep, 0, 1);
                __builtin_prefetch(pu + bstep, 0, 1);
            }
        }

        // 2) fragments + 8 WMMAs from current buffer
        v16bf a[2], bg[2], bu[2];
#pragma unroll
        for (int i = 0; i < 2; ++i) {
            const __bf16* p = &lA[cur][wr * 32 + i * 16 + lr][aoff];
            a[i] = frag_from(p, p + 16);
        }
#pragma unroll
        for (int j = 0; j < 2; ++j) {
            const __bf16* pg2 = &lBg[cur][wc * 32 + j * 16 + lr][boff];
            const __bf16* pu2 = &lBu[cur][wc * 32 + j * 16 + lr][boff];
            bg[j] = frag_from(pg2, pg2 + 16);
            bu[j] = frag_from(pu2, pu2 + 16);
        }
#pragma unroll
        for (int i = 0; i < 2; ++i)
#pragma unroll
            for (int j = 0; j < 2; ++j) {
                accG[i][j] = __builtin_amdgcn_wmma_f32_16x16x32_bf16(
                    false, a[i], false, bg[j], (short)0, accG[i][j], false, false);
                accU[i][j] = __builtin_amdgcn_wmma_f32_16x16x32_bf16(
                    false, a[i], false, bu[j], (short)0, accU[i][j], false, false);
            }

        // 3) stage next step into the other buffer
        if (has_next) {
            const int nxt = cur ^ 1;
            uint4 w0 = {pk2(a0.x,a0.y), pk2(a0.z,a0.w), pk2(a1.x,a1.y), pk2(a1.z,a1.w)};
            uint4 w1 = {pk2(a2.x,a2.y), pk2(a2.z,a2.w), pk2(a3.x,a3.y), pk2(a3.z,a3.w)};
            *(uint4*)&lA[nxt][arow][acb + 0] = w0;
            *(uint4*)&lA[nxt][arow][acb + 8] = w1;
            const int kk = 2 * bkp;
            *(unsigned*)&lBg[nxt][bnb + 0][kk] = pk2(g0.x, g1.x);
            *(unsigned*)&lBg[nxt][bnb + 1][kk] = pk2(g0.y, g1.y);
            *(unsigned*)&lBg[nxt][bnb + 2][kk] = pk2(g0.z, g1.z);
            *(unsigned*)&lBg[nxt][bnb + 3][kk] = pk2(g0.w, g1.w);
            *(unsigned*)&lBu[nxt][bnb + 0][kk] = pk2(u0.x, u1.x);
            *(unsigned*)&lBu[nxt][bnb + 1][kk] = pk2(u0.y, u1.y);
            *(unsigned*)&lBu[nxt][bnb + 2][kk] = pk2(u0.z, u1.z);
            *(unsigned*)&lBu[nxt][bnb + 3][kk] = pk2(u0.w, u1.w);
        }
        __syncthreads();
    }

    // ---- SwiGLU epilogue: hidden = up * gate * sigmoid(gate) -> bf16 ----
    __bf16* He = Hid + (size_t)e * TE * F;
    const int moff = (lane < 16) ? 0 : 8;
    const int ncol = lane & 15;
#pragma unroll
    for (int i = 0; i < 2; ++i)
#pragma unroll
        for (int j = 0; j < 2; ++j) {
            const int mb = m0 + wr * 32 + i * 16 + moff;
            const int nb = n0 + wc * 32 + j * 16 + ncol;
#pragma unroll
            for (int r = 0; r < 8; ++r) {
                float g = accG[i][j][r];
                float u = accU[i][j][r];
                float h = u * (g / (1.0f + __expf(-g)));
                He[(size_t)(mb + r) * F + nb] = (__bf16)h;
            }
        }
}

// ---------------------------------------------------------------------------
// Kernel 2: down projection
//   Hid : [E][TE][F] bf16 ; Wd : [E][F][H] fp32 ; Out : [E][TE][H] fp32
// A-tile staged with GLOBAL_LOAD_ASYNC_TO_LDS_B128 (no VGPR round-trip).
// ---------------------------------------------------------------------------
__global__ __launch_bounds__(256) void
moe_down(const __bf16* __restrict__ Hid,
         const float* __restrict__ Wd,
         float* __restrict__ Out) {
    __shared__ __bf16 lA[2][BM][LDA];
    __shared__ __bf16 lB[2][BN][LDA];

    const int e  = blockIdx.z;
    const int m0 = blockIdx.y * BM;
    const int n0 = blockIdx.x * BN;          // column offset in H-space

    const __bf16* Ae = Hid + (size_t)e * TE * F;
    const float*  We = Wd  + (size_t)e * F * H;

    const int tid  = threadIdx.x;
    const int lane = tid & 31;
    const int w    = tid >> 5;
    const int wr   = w >> 1;
    const int wc   = w & 1;

    const v8f zero = {0.f,0.f,0.f,0.f,0.f,0.f,0.f,0.f};
    v8f acc[2][2] = {{zero, zero}, {zero, zero}};

    const int lr   = lane & 15;
    const int sel  = lane >> 4;
    const int aoff = sel * 8;
    const int boff = sel * 8;

    const int arow = tid >> 1;               // A: 16 bf16 (32 B) per thread
    const int acb  = (tid & 1) * 16;
    const int bkp  = tid >> 4;               // B: K-pair, 4 columns per thread
    const int bnb  = (tid & 15) * 4;

    const __bf16* srcA = Ae + (size_t)(m0 + arow) * F + acb;
    const float*  srcB = We + (size_t)(2 * bkp) * H + n0 + bnb;
    const size_t  bstep = (size_t)BK * H;

    // per-lane LDS byte offsets for the async A copies
    const unsigned ldsA0[2] = { lds_off(&lA[0][arow][acb]),
                                lds_off(&lA[1][arow][acb]) };

    // ---- prologue: async A + register-path B into buffer 0 ----
    {
        async_b128(ldsA0[0] + 0,  srcA + 0);
        async_b128(ldsA0[0] + 16, srcA + 8);
        float4 b0 = *(const float4*)(srcB);
        float4 b1 = *(const float4*)(srcB + H);
        const int kk = 2 * bkp;
        *(unsigned*)&lB[0][bnb + 0][kk] = pk2(b0.x, b1.x);
        *(unsigned*)&lB[0][bnb + 1][kk] = pk2(b0.y, b1.y);
        *(unsigned*)&lB[0][bnb + 2][kk] = pk2(b0.z, b1.z);
        *(unsigned*)&lB[0][bnb + 3][kk] = pk2(b0.w, b1.w);
        wait_async0();
    }
    __syncthreads();

    constexpr int NSTEP = F / BK;            // 128
    for (int s = 0; s < NSTEP; ++s) {
        const int cur = s & 1;
        const bool has_next = (s + 1) < NSTEP;
        const int nxt = cur ^ 1;

        // 1) issue next-step copies early (overlap with WMMA below)
        float4 b0, b1;
        if (has_next) {
            const __bf16* pa = srcA + (s + 1) * BK;
            const float*  pb = srcB + (s + 1) * bstep;
            async_b128(ldsA0[nxt] + 0,  pa + 0);   // A: straight bf16, no VGPRs
            async_b128(ldsA0[nxt] + 16, pa + 8);
            b0 = *(const float4*)(pb);             // B: needs fp32->bf16
            b1 = *(const float4*)(pb + H);
            if (s + 2 < NSTEP) {
                __builtin_prefetch(pa + BK, 0, 1);
                __builtin_prefetch(pb + bstep, 0, 1);
            }
        }

        // 2) fragments + 4 WMMAs from current buffer
        v16bf a[2], b[2];
#pragma unroll
        for (int i = 0; i < 2; ++i) {
            const __bf16* p = &lA[cur][wr * 32 + i * 16 + lr][aoff];
            a[i] = frag_from(p, p + 16);
        }
#pragma unroll
        for (int j = 0; j < 2; ++j) {
            const __bf16* p = &lB[cur][wc * 32 + j * 16 + lr][boff];
            b[j] = frag_from(p, p + 16);
        }
#pragma unroll
        for (int i = 0; i < 2; ++i)
#pragma unroll
            for (int j = 0; j < 2; ++j)
                acc[i][j] = __builtin_amdgcn_wmma_f32_16x16x32_bf16(
                    false, a[i], false, b[j], (short)0, acc[i][j], false, false);

        // 3) publish next buffer
        if (has_next) {
            const int kk = 2 * bkp;
            *(unsigned*)&lB[nxt][bnb + 0][kk] = pk2(b0.x, b1.x);
            *(unsigned*)&lB[nxt][bnb + 1][kk] = pk2(b0.y, b1.y);
            *(unsigned*)&lB[nxt][bnb + 2][kk] = pk2(b0.z, b1.z);
            *(unsigned*)&lB[nxt][bnb + 3][kk] = pk2(b0.w, b1.w);
            wait_async0();                       // A copies landed in LDS
        }
        __syncthreads();
    }

    // ---- write fp32 output ----
    float* Oe = Out + (size_t)e * TE * H;
    const int moff = (lane < 16) ? 0 : 8;
    const int ncol = lane & 15;
#pragma unroll
    for (int i = 0; i < 2; ++i)
#pragma unroll
        for (int j = 0; j < 2; ++j) {
            const int mb = m0 + wr * 32 + i * 16 + moff;
            const int nb = n0 + wc * 32 + j * 16 + ncol;
#pragma unroll
            for (int r = 0; r < 8; ++r)
                Oe[(size_t)(mb + r) * H + nb] = acc[i][j][r];
        }
}

// ---------------------------------------------------------------------------
extern "C" void kernel_launch(void* const* d_in, const int* in_sizes, int n_in,
                              void* d_out, int out_size, void* d_ws, size_t ws_size,
                              hipStream_t stream) {
    const float* X   = (const float*)d_in[0];   // [T,H] fp32 (pre-grouped)
    const float* Wgu = (const float*)d_in[1];   // [E,H,2F] fp32
    const float* Wd  = (const float*)d_in[2];   // [E,F,H] fp32
    float*   out = (float*)d_out;               // [T,H] fp32
    __bf16*  hid = (__bf16*)d_ws;               // [E,TE,F] bf16 (64 MiB)

    dim3 blk(256);
    // blockIdx.z = expert so each expert's weight slab (<=67 MB) stays L2-hot.
    moe_gateup_swiglu<<<dim3(F / BN, TE / BM, E), blk, 0, stream>>>(X, Wgu, hid);
    moe_down        <<<dim3(H / BN, TE / BM, E), blk, 0, stream>>>(hid, Wd, out);
}
